// GraphPooling_57071525430035
// MI455X (gfx1250) — matrix operand order repaired and checked by reading, not compiled, not run on hardware
//
#include <hip/hip_runtime.h>
#include <hip/hip_bf16.h>

// ---------------- problem constants ----------------
#define NB   64           // graphs
#define NPG  512          // nodes per graph
#define NN   (NB*NPG)     // 32768 nodes
#define NE   (NN*16)      // 524288 edges
#define INF_ 128
#define HH   128
#define H3   384
#define K1   256
#define K2   128
#define N1v  (NB*K1)      // 16384
#define N2v  (NB*K2)      // 8192
#define E1v  (NE+NN)      // 557056 (edges + self loops, stage 1)
#define E2v  (NE+N1v)     // 540672 (edges + self loops, stage 2)
#define EPSV 1e-5f
#define CDIV(a,b) (((a)+(b)-1)/(b))

typedef _Float16 h16v __attribute__((ext_vector_type(16)));
typedef _Float16 h8v  __attribute__((ext_vector_type(8)));
typedef float    f8v  __attribute__((ext_vector_type(8)));

// ---------------- small utility kernels ----------------
__global__ void k_fillf(float* p, float v, int n){ int i=blockIdx.x*blockDim.x+threadIdx.x; if(i<n) p[i]=v; }
__global__ void k_filli(int* p, int v, int n){ int i=blockIdx.x*blockDim.x+threadIdx.x; if(i<n) p[i]=v; }
__global__ void k_fillu(unsigned* p, unsigned v, int n){ int i=blockIdx.x*blockDim.x+threadIdx.x; if(i<n) p[i]=v; }
__global__ void k_copyi(const int* __restrict__ s, int* __restrict__ d, int n){
  int i=blockIdx.x*blockDim.x+threadIdx.x; if(i<n) d[i]=s[i]; }
__global__ void k_iota_div(float* p, int dv, int n){
  int i=blockIdx.x*blockDim.x+threadIdx.x; if(i<n) p[i]=(float)(i/dv); }
__global__ void k_f2h(const float* __restrict__ s, _Float16* __restrict__ d, int n){
  int i=blockIdx.x*blockDim.x+threadIdx.x; if(i<n) d[i]=(_Float16)s[i]; }

// ---------------- WMMA GEMM:  Y[M,N] = X[M,K] @ W[N,K]^T  (f16 in, f32 out) ----------------
// One wave per 16x16 output tile; K stepped by 32 using v_wmma_f32_16x16x32_f16.
__global__ void k_gemm(const _Float16* __restrict__ X, const _Float16* __restrict__ W,
                       float* __restrict__ Y, int M, int Nout, int K){
  int wave = blockIdx.x*(blockDim.x>>5) + (threadIdx.x>>5);
  int lane = threadIdx.x & 31;
  int tilesN = Nout >> 4;
  int tiles  = (M>>4)*tilesN;
  if (wave >= tiles) return;                 // wave-uniform branch: EXEC all-1 inside
  int tm = wave / tilesN, tn = wave % tilesN;
  int half = lane >> 4, l16 = lane & 15;
  const _Float16* xrow = X + (size_t)(tm*16 + l16)*K;
  const _Float16* wrow = W + (size_t)(tn*16 + l16)*K;
  f8v c = {0.f,0.f,0.f,0.f,0.f,0.f,0.f,0.f};
  for (int k0 = 0; k0 < K; k0 += 32){
    // A fragment (16x32, M=l16): elems 0..7 -> K=k0+8*half+e ; elems 8..15 -> K=k0+16+8*half+(e-8)
    h8v alo = *(const h8v*)(xrow + k0 + half*8);
    h8v ahi = *(const h8v*)(xrow + k0 + 16 + half*8);
    h16v a, b;
    // B fragment (32x16, N=l16): elems 0..7 -> K=k0+8*half+e ; elems 8..15 -> K=k0+16+8*half+(e-8)
    h8v blo = *(const h8v*)(wrow + k0 + half*8);
    h8v bhi = *(const h8v*)(wrow + k0 + 16 + half*8);
#pragma unroll
    for (int i=0;i<8;i++){ a[i]=alo[i]; a[8+i]=ahi[i]; b[i]=blo[i]; b[8+i]=bhi[i]; }
    c = __builtin_amdgcn_wmma_f32_16x16x32_f16(false, a, false, b, (short)0, c, false, false);
  }
  int col = tn*16 + l16;
#pragma unroll
  for (int r=0;r<8;r++){
    int row = tm*16 + half*8 + r;            // C/D layout: VGPR r = row 8*half + r
    Y[(size_t)row*Nout + col] = c[r];
  }
}

// ---------------- per-node dots with attention vectors ----------------
__global__ void k_rowdot2(const float* __restrict__ xp, const float* __restrict__ a_src,
                          const float* __restrict__ a_dst, float* __restrict__ osrc,
                          float* __restrict__ odst, int n, int K){
  int node = blockIdx.x*(blockDim.x>>5) + (threadIdx.x>>5);
  int lane = threadIdx.x & 31;
  if (node >= n) return;
  const float* row = xp + (size_t)node*K;
  float s1=0.f, s2=0.f;
  for (int k=lane;k<K;k+=32){ float v=row[k]; s1+=v*a_src[k]; s2+=v*a_dst[k]; }
  for (int off=16; off; off>>=1){ s1 += __shfl_down(s1, off, 32); s2 += __shfl_down(s2, off, 32); }
  if (!lane){ osrc[node]=s1; odst[node]=s2; }
}

// ---------------- edge softmax (segment max/sum via atomics) ----------------
__device__ __forceinline__ unsigned fenc(float f){
  unsigned u=__float_as_uint(f); return (u&0x80000000u)? ~u : (u|0x80000000u); }
__device__ __forceinline__ float fdec(unsigned u){
  return (u&0x80000000u)? __uint_as_float(u&0x7fffffffu) : __uint_as_float(~u); }

__global__ void k_edge_score(const int* __restrict__ s, const int* __restrict__ d,
                             const int* __restrict__ mask, const float* __restrict__ asrc,
                             const float* __restrict__ adst, float* __restrict__ e_out,
                             unsigned* __restrict__ menc, int ne){
  int i=blockIdx.x*blockDim.x+threadIdx.x; if(i>=ne) return;
  float e;
  if (mask && !mask[i]) e = -1e9f;
  else { float t = asrc[s[i]] + adst[d[i]]; e = (t>0.f)? t : 0.2f*t; }   // leaky_relu(0.2)
  e_out[i]=e;
  atomicMax(menc + d[i], fenc(e));
}
__global__ void k_expsum(const int* __restrict__ d, const int* __restrict__ mask,
                         float* __restrict__ ealpha, const unsigned* __restrict__ menc,
                         float* __restrict__ ssum, int ne){
  int i=blockIdx.x*blockDim.x+threadIdx.x; if(i>=ne) return;
  float ex = 0.f;
  if (!mask || mask[i]) ex = expf(ealpha[i] - fdec(menc[d[i]]));
  ealpha[i]=ex;
  atomicAdd(ssum + d[i], ex);
}
__global__ void k_norm(const int* __restrict__ d, float* __restrict__ alpha,
                       const float* __restrict__ ssum, int ne){
  int i=blockIdx.x*blockDim.x+threadIdx.x; if(i>=ne) return;
  alpha[i] = alpha[i] / ssum[d[i]];
}

// ---------------- dense-adjacency accumulation (fea1/fea2) ----------------
__global__ void k_fea(const int* __restrict__ s, const int* __restrict__ d,
                      const float* __restrict__ alpha, int ne, int nper,
                      float* __restrict__ fea, float scale){
  int i=blockIdx.x*blockDim.x+threadIdx.x; if(i>=ne) return;
  int sv=s[i]; int g=sv/nper; int u=sv-g*nper; int v=d[i]-g*nper;
  atomicAdd(&fea[u*nper+v], alpha[i]*scale);
}

// ---------------- CSR build (incoming edges per dst node) ----------------
__global__ void k_count(const int* __restrict__ d, int* __restrict__ deg, int ne){
  int i=blockIdx.x*blockDim.x+threadIdx.x; if(i<ne) atomicAdd(&deg[d[i]],1); }
__global__ void k_scan(const int* __restrict__ in, int* __restrict__ out, int n){
  __shared__ int sh[1024];
  int tid = threadIdx.x;
  int per = (n + 1023) >> 10;
  int beg = tid*per; int end = beg+per; if (end>n) end=n; if (beg>n) beg=n;
  int sum=0; for (int i=beg;i<end;i++) sum+=in[i];
  sh[tid]=sum; __syncthreads();
  for (int off=1; off<1024; off<<=1){
    int t = (tid>=off)? sh[tid-off] : 0;
    __syncthreads();
    sh[tid]+=t;
    __syncthreads();
  }
  int run = sh[tid]-sum;
  for (int i=beg;i<end;i++){ out[i]=run; run+=in[i]; }
  if (tid==1023) out[n]=sh[1023];
}
__global__ void k_fill_csr(const int* __restrict__ d, int* __restrict__ cursor,
                           int* __restrict__ eid, int ne){
  int i=blockIdx.x*blockDim.x+threadIdx.x; if(i>=ne) return;
  int pos = atomicAdd(&cursor[d[i]],1);
  eid[pos]=i;
}

// ---------------- gather aggregation: out[n] = sum_e alpha[e]*xp[src_e] + bias ----------------
__global__ void k_aggregate(const float* __restrict__ xp, const float* __restrict__ alpha,
                            const int* __restrict__ sArr, const int* __restrict__ rowstart,
                            const int* __restrict__ eid, const float* __restrict__ bias,
                            float* __restrict__ out, int n, int ldo){
  int node = blockIdx.x; int c = threadIdx.x;   // blockDim = 128 channels
  if (node >= n) return;
  float acc = bias[c];
  int b0 = rowstart[node], b1 = rowstart[node+1];
  for (int p=b0;p<b1;p++){
    int e = eid[p];
    acc += alpha[e] * xp[(size_t)sArr[e]*HH + c];
  }
  out[(size_t)node*ldo + c] = acc;
}

// ---------------- BatchNorm over nodes (biased var), optional pre-ReLU ----------------
__global__ void k_bnstat(const float* __restrict__ x, int n, int ld, int relu,
                         float* __restrict__ sums){
  int c = threadIdx.x;                       // 128 channels
  int r0 = blockIdx.x*256, r1 = r0+256; if (r1>n) r1=n;
  float s=0.f, q=0.f;
  for (int r=r0;r<r1;r++){ float v=x[(size_t)r*ld+c]; if (relu) v=fmaxf(v,0.f); s+=v; q+=v*v; }
  atomicAdd(&sums[c], s); atomicAdd(&sums[128+c], q);
}
__global__ void k_bnapply(float* __restrict__ x, int n, int ld, int relu,
                          const float* __restrict__ sums, const float* __restrict__ w,
                          const float* __restrict__ b, _Float16* __restrict__ xh, int ldh){
  int idx = blockIdx.x*blockDim.x+threadIdx.x; if (idx >= n*HH) return;
  int r = idx>>7, c = idx&127;
  float mu = sums[c]/(float)n;
  float var = sums[128+c]/(float)n - mu*mu;
  float v = x[(size_t)r*ld+c]; if (relu) v=fmaxf(v,0.f);
  float y = (v-mu)*(w[c]/sqrtf(var+EPSV)) + b[c];
  x[(size_t)r*ld+c]=y;
  if (xh) xh[(size_t)r*ldh+c]=(_Float16)y;
}

// ---------------- TopK pooling ----------------
__global__ void k_vnorm(const float* __restrict__ w, int n, float* __restrict__ out){
  __shared__ float sh[512];
  int t=threadIdx.x; float s=0.f;
  for (int i=t;i<n;i+=512) s+=w[i]*w[i];
  sh[t]=s; __syncthreads();
  for (int off=256; off; off>>=1){ if (t<off) sh[t]+=sh[t+off]; __syncthreads(); }
  if (!t) out[0]=sqrtf(sh[0]);
}
__global__ void k_score(const float* __restrict__ x, const float* __restrict__ w,
                        const float* __restrict__ wnorm, float* __restrict__ sc, int n){
  int node = blockIdx.x*(blockDim.x>>5) + (threadIdx.x>>5);
  int lane = threadIdx.x & 31;
  if (node >= n) return;
  const float* row = x + (size_t)node*H3;
  float s=0.f;
  for (int k=lane;k<H3;k+=32) s += row[k]*w[k];
  for (int off=16; off; off>>=1) s += __shfl_down(s, off, 32);
  if (!lane) sc[node] = tanhf(s/wnorm[0]);
}
// bitonic sort per graph, descending value / ascending index tie-break (jax.lax.top_k order)
template<int NNODES, int KSEL>
__global__ void k_topk(const float* __restrict__ sc, int* __restrict__ perm_i,
                       float* __restrict__ perm_f, float* __restrict__ score_f){
  __shared__ float sv[NNODES]; __shared__ int si[NNODES];
  int g = blockIdx.x, t = threadIdx.x;
  sv[t] = sc[g*NNODES + t]; si[t] = t; __syncthreads();
  for (int k=2;k<=NNODES;k<<=1)
    for (int j=k>>1;j>0;j>>=1){
      int p = t ^ j;
      if (p > t){
        float va=sv[t], vb=sv[p]; int ia=si[t], ib=si[p];
        bool up = ((t & k) == 0);
        bool less = (va>vb) || (va==vb && ia<ib);   // "a comes first" in descending order
        if (up ? !less : less){ sv[t]=vb; si[t]=ib; sv[p]=va; si[p]=ia; }
      }
      __syncthreads();
    }
  if (t < KSEL){
    int gi = g*KSEL + t;
    int gp = g*NNODES + si[t];
    perm_i[gi]  = gp;
    perm_f[gi]  = (float)gp;
    score_f[gi] = sv[t];
  }
}
__global__ void k_pool_gather(const float* __restrict__ x, const int* __restrict__ perm,
                              const float* __restrict__ sc, float* __restrict__ out){
  int i = blockIdx.x; int c = threadIdx.x;   // 384
  out[(size_t)i*H3+c] = x[(size_t)perm[i]*H3+c] * sc[i];
}
__global__ void k_scatter_map(const int* __restrict__ perm, int* __restrict__ nmap, int n){
  int i=blockIdx.x*blockDim.x+threadIdx.x; if(i<n) nmap[perm[i]]=i; }

// ---------------- GraphNorm ----------------
__global__ void k_gnorm(const float* __restrict__ x, const float* __restrict__ wB,
                        const float* __restrict__ bB, const float* __restrict__ msB,
                        int kper, float* __restrict__ of32, _Float16* __restrict__ oh){
  int g = blockIdx.x, c = threadIdx.x;   // 384 threads
  const float* base = x + (size_t)g*kper*H3 + c;
  float mean=0.f; for (int r=0;r<kper;r++) mean += base[(size_t)r*H3];
  mean /= (float)kper;
  float mm = mean*msB[c];
  float var=0.f; for (int r=0;r<kper;r++){ float t=base[(size_t)r*H3]-mm; var+=t*t; }
  var /= (float)kper;
  float sc = wB[c]/sqrtf(var+EPSV); float bb=bB[c];
  for (int r=0;r<kper;r++){
    float y = (base[(size_t)r*H3]-mm)*sc + bb;
    size_t o = ((size_t)(g*kper+r))*H3 + c;
    if (of32) of32[o]=y;
    if (oh)   oh[o]=(_Float16)y;
  }
}

// ---------------- edge-list construction ----------------
__global__ void k_build_e1(const int* __restrict__ ei, int* __restrict__ s1, int* __restrict__ d1){
  int i=blockIdx.x*blockDim.x+threadIdx.x; if(i>=E1v) return;
  if (i<NE){ s1[i]=ei[i]; d1[i]=ei[NE+i]; } else { s1[i]=i-NE; d1[i]=i-NE; }
}
__global__ void k_build_e2(const int* __restrict__ ei, const int* __restrict__ nmap,
                           int* __restrict__ s2, int* __restrict__ d2, int* __restrict__ m2){
  int i=blockIdx.x*blockDim.x+threadIdx.x; if(i>=E2v) return;
  if (i<NE){
    int ns=nmap[ei[i]], nd=nmap[ei[NE+i]];
    int v = (ns>=0 && nd>=0) ? 1 : 0;
    s2[i]= v?ns:0; d2[i]= v?nd:0; m2[i]=v;
  } else { int l=i-NE; s2[i]=l; d2[i]=l; m2[i]=1; }
}

// ---------------- readout + MLP ----------------
__global__ void k_gmean(const float* __restrict__ xq, float* __restrict__ xg){
  int g=blockIdx.x, c=threadIdx.x;           // 384
  float s=0.f; for (int r=0;r<K2;r++) s += xq[((size_t)(g*K2+r))*H3+c];
  xg[g*H3+c] = s/(float)K2;
}
__global__ void k_linrelu(const float* __restrict__ X, const float* __restrict__ W,
                          const float* __restrict__ b, float* __restrict__ Y,
                          int M, int Nout, int K){
  int idx=blockIdx.x*blockDim.x+threadIdx.x; if(idx>=M*Nout) return;
  int m=idx/Nout, n=idx%Nout;
  const float* xr=X+(size_t)m*K; const float* wr=W+(size_t)n*K;
  float s=b[n];
  for (int k=0;k<K;k++) s += xr[k]*wr[k];
  Y[idx]=fmaxf(s,0.f);
}
__global__ void k_bnsmall(const float* __restrict__ x, const float* __restrict__ w,
                          const float* __restrict__ b, float* __restrict__ y, int M, int Nc){
  int c=blockIdx.x*blockDim.x+threadIdx.x; if(c>=Nc) return;
  float s=0.f,q=0.f;
  for (int r=0;r<M;r++){ float v=x[(size_t)r*Nc+c]; s+=v; q+=v*v; }
  float mu=s/(float)M, var=q/(float)M-mu*mu;
  float sc=w[c]/sqrtf(var+EPSV);
  for (int r=0;r<M;r++) y[(size_t)r*Nc+c]=(x[(size_t)r*Nc+c]-mu)*sc+b[c];
}

// ---------------- one GAT layer (host helper) ----------------
static void gat_layer(hipStream_t st, const _Float16* xh, int Kdim, int nnode, int ne,
                      const int* sA, const int* dA, const int* mA,
                      const float* Wp, const float* a_src, const float* a_dst, const float* bias,
                      int relu, const float* bnw, const float* bnb,
                      _Float16* wh, float* xp, float* asrc, float* adst,
                      unsigned* menc, float* ssum, float* alpha,
                      const int* rowstart, const int* eid, float* bnsum,
                      float* out_slice, _Float16* xh_next,
                      float* fea, int nper, float fscale){
  k_f2h<<<CDIV(HH*Kdim,256),256,0,st>>>(Wp, wh, HH*Kdim);
  int tiles = (nnode/16)*(HH/16);
  k_gemm<<<CDIV(tiles,8),256,0,st>>>(xh, wh, xp, nnode, HH, Kdim);
  k_rowdot2<<<CDIV(nnode,8),256,0,st>>>(xp, a_src, a_dst, asrc, adst, nnode, HH);
  k_fillu<<<CDIV(nnode,256),256,0,st>>>(menc, 0u, nnode);
  k_fillf<<<CDIV(nnode,256),256,0,st>>>(ssum, 0.f, nnode);
  k_edge_score<<<CDIV(ne,256),256,0,st>>>(sA, dA, mA, asrc, adst, alpha, menc, ne);
  k_expsum<<<CDIV(ne,256),256,0,st>>>(dA, mA, alpha, menc, ssum, ne);
  k_norm<<<CDIV(ne,256),256,0,st>>>(dA, alpha, ssum, ne);
  k_fea<<<CDIV(ne,256),256,0,st>>>(sA, dA, alpha, ne, nper, fea, fscale);
  k_aggregate<<<nnode,128,0,st>>>(xp, alpha, sA, rowstart, eid, bias, out_slice, nnode, H3);
  k_fillf<<<1,256,0,st>>>(bnsum, 0.f, 256);
  k_bnstat<<<CDIV(nnode,256),128,0,st>>>(out_slice, nnode, H3, relu, bnsum);
  k_bnapply<<<CDIV(nnode*HH,256),256,0,st>>>(out_slice, nnode, H3, relu, bnsum, bnw, bnb, xh_next, HH);
}

// ---------------- entry point ----------------
extern "C" void kernel_launch(void* const* d_in, const int* in_sizes, int n_in,
                              void* d_out, int out_size, void* d_ws, size_t ws_size,
                              hipStream_t stream) {
  (void)in_sizes; (void)n_in; (void)out_size; (void)ws_size;
  // ---- input mapping: setup_inputs() insertion order; params jax-flattened (sorted keys) ----
  // 0:x 1:edge_attr(unused)
  // gc{1,2}: bn0.b bn0.w bn1.b bn1.w bn2.b bn2.w | gat{0,1,2}: W a_dst a_src b
  // gn{1,2}: b ms w | mlp: W1 W2 b1 b2 bn1.b bn1.w bn2.b bn2.w | pool1_w pool2_w
  // then edge_index, batch(unused)
  const float* x = (const float*)d_in[0];
  const float *gc_bn_b[2][3], *gc_bn_w[2][3], *gc_W[2][3], *gc_ad[2][3], *gc_as[2][3], *gc_b[2][3];
  int pi = 2;
  for (int s=0;s<2;s++){
    for (int l=0;l<3;l++){ gc_bn_b[s][l]=(const float*)d_in[pi++]; gc_bn_w[s][l]=(const float*)d_in[pi++]; }
    for (int l=0;l<3;l++){ gc_W[s][l]=(const float*)d_in[pi++]; gc_ad[s][l]=(const float*)d_in[pi++];
                           gc_as[s][l]=(const float*)d_in[pi++]; gc_b[s][l]=(const float*)d_in[pi++]; }
  }
  const float *gn_b[2], *gn_ms[2], *gn_w[2];
  for (int s=0;s<2;s++){ gn_b[s]=(const float*)d_in[pi++]; gn_ms[s]=(const float*)d_in[pi++]; gn_w[s]=(const float*)d_in[pi++]; }
  const float* W1=(const float*)d_in[pi++]; const float* W2=(const float*)d_in[pi++];
  const float* b1=(const float*)d_in[pi++]; const float* b2=(const float*)d_in[pi++];
  const float* bn1b=(const float*)d_in[pi++]; const float* bn1w=(const float*)d_in[pi++];
  const float* bn2b=(const float*)d_in[pi++]; const float* bn2w=(const float*)d_in[pi++];
  const float* pw1=(const float*)d_in[pi++]; const float* pw2=(const float*)d_in[pi++];
  const int* edge_index=(const int*)d_in[pi++];

  // ---- output layout (flat float, return-tuple order) ----
  float* outp = (float*)d_out;
  float* o_h      = outp;             // 64*256
  float* o_perm1  = outp + 16384;     // 16384
  float* o_perm2  = outp + 32768;     // 8192
  float* o_score1 = outp + 40960;     // 16384
  float* o_score2 = outp + 57344;     // 8192
  float* o_batch1 = outp + 65536;     // 16384
  float* o_batch2 = outp + 81920;     // 8192
  float* o_fea1   = outp + 90112;     // 512*512
  float* o_fea2   = outp + 352256;    // 256*256

  // ---- workspace carve (256B aligned) ----
  char* wsp = (char*)d_ws;
  auto alloc = [&](size_t bytes)->void*{ void* p=(void*)wsp; wsp += (bytes + 255) & ~(size_t)255; return p; };
  _Float16* xh_a = (_Float16*)alloc((size_t)N1v*H3*sizeof(_Float16)); // stage-2 384-wide f16 input
  _Float16* xh_b = (_Float16*)alloc((size_t)NN*HH*sizeof(_Float16));  // 128-wide f16 activations
  _Float16* wh   = (_Float16*)alloc((size_t)HH*H3*sizeof(_Float16)); // current layer weights (f16)
  float* xp      = (float*)alloc((size_t)NN*HH*sizeof(float));       // x @ W^T
  float* asrc    = (float*)alloc((size_t)NN*sizeof(float));
  float* adst    = (float*)alloc((size_t)NN*sizeof(float));
  unsigned* menc = (unsigned*)alloc((size_t)NN*sizeof(unsigned));
  float* ssum    = (float*)alloc((size_t)NN*sizeof(float));
  float* alpha   = (float*)alloc((size_t)E1v*sizeof(float));
  int* s1 = (int*)alloc((size_t)E1v*sizeof(int));
  int* d1 = (int*)alloc((size_t)E1v*sizeof(int));
  int* s2 = (int*)alloc((size_t)E2v*sizeof(int));
  int* d2 = (int*)alloc((size_t)E2v*sizeof(int));
  int* m2 = (int*)alloc((size_t)E2v*sizeof(int));
  int* deg      = (int*)alloc((size_t)(NN+1)*sizeof(int));
  int* rowstart = (int*)alloc((size_t)(NN+1)*sizeof(int));
  int* cursor   = (int*)alloc((size_t)(NN+1)*sizeof(int));
  int* eid      = (int*)alloc((size_t)E1v*sizeof(int));
  float* bnsum  = (float*)alloc(256*sizeof(float));
  float* xall   = (float*)alloc((size_t)NN*H3*sizeof(float));  // concat features (reused stage-2)
  float* xp1    = (float*)alloc((size_t)N1v*H3*sizeof(float)); // pooled (reused for pool-2)
  float* score_all = (float*)alloc((size_t)NN*sizeof(float));
  int* perm1_i  = (int*)alloc((size_t)N1v*sizeof(int));
  int* perm2_i  = (int*)alloc((size_t)N2v*sizeof(int));
  int* nmap     = (int*)alloc((size_t)NN*sizeof(int));
  float* xq     = (float*)alloc((size_t)N2v*H3*sizeof(float));
  float* xg     = (float*)alloc((size_t)NB*H3*sizeof(float));
  float* h1     = (float*)alloc((size_t)64*512*sizeof(float));
  float* h2     = (float*)alloc((size_t)64*256*sizeof(float));
  float* wnorm  = (float*)alloc(256);

  const float fsc = 1.f/(3.f*(float)NB);

  // ---- init outputs that are accumulated / trivially computed ----
  k_fillf<<<CDIV(NPG*NPG,256),256,0,stream>>>(o_fea1, 0.f, NPG*NPG);
  k_fillf<<<CDIV(K1*K1,256),256,0,stream>>>(o_fea2, 0.f, K1*K1);
  k_iota_div<<<CDIV(N1v,256),256,0,stream>>>(o_batch1, K1, N1v);
  k_iota_div<<<CDIV(N2v,256),256,0,stream>>>(o_batch2, K2, N2v);

  // ---- stage 1: edges + CSR ----
  k_build_e1<<<CDIV(E1v,256),256,0,stream>>>(edge_index, s1, d1);
  k_filli<<<CDIV(NN,256),256,0,stream>>>(deg, 0, NN);
  k_count<<<CDIV(E1v,256),256,0,stream>>>(d1, deg, E1v);
  k_scan<<<1,1024,0,stream>>>(deg, rowstart, NN);
  k_copyi<<<CDIV(NN,256),256,0,stream>>>(rowstart, cursor, NN);
  k_fill_csr<<<CDIV(E1v,256),256,0,stream>>>(d1, cursor, eid, E1v);

  // ---- stage 1 GAT stack (relu on layers 0 and 2, not on layer 1) ----
  k_f2h<<<CDIV(NN*HH,256),256,0,stream>>>(x, xh_b, NN*HH);
  const int relu1[3] = {1,0,1};
  for (int l=0;l<3;l++)
    gat_layer(stream, xh_b, HH, NN, E1v, s1, d1, nullptr,
              gc_W[0][l], gc_as[0][l], gc_ad[0][l], gc_b[0][l],
              relu1[l], gc_bn_w[0][l], gc_bn_b[0][l],
              wh, xp, asrc, adst, menc, ssum, alpha, rowstart, eid, bnsum,
              xall + l*HH, xh_b, o_fea1, NPG, fsc);

  // ---- TopK pool 1 + graph norm ----
  k_vnorm<<<1,512,0,stream>>>(pw1, H3, wnorm);
  k_score<<<CDIV(NN,8),256,0,stream>>>(xall, pw1, wnorm, score_all, NN);
  k_topk<NPG,K1><<<NB,NPG,0,stream>>>(score_all, perm1_i, o_perm1, o_score1);
  k_pool_gather<<<N1v,H3,0,stream>>>(xall, perm1_i, o_score1, xp1);
  k_filli<<<CDIV(NN,256),256,0,stream>>>(nmap, -1, NN);
  k_scatter_map<<<CDIV(N1v,256),256,0,stream>>>(perm1_i, nmap, N1v);
  k_gnorm<<<NB,H3,0,stream>>>(xp1, gn_w[0], gn_b[0], gn_ms[0], K1, nullptr, xh_a);

  // ---- stage 2: remapped edges + CSR ----
  k_build_e2<<<CDIV(E2v,256),256,0,stream>>>(edge_index, nmap, s2, d2, m2);
  k_filli<<<CDIV(N1v,256),256,0,stream>>>(deg, 0, N1v);
  k_count<<<CDIV(E2v,256),256,0,stream>>>(d2, deg, E2v);
  k_scan<<<1,1024,0,stream>>>(deg, rowstart, N1v);
  k_copyi<<<CDIV(N1v,256),256,0,stream>>>(rowstart, cursor, N1v);
  k_fill_csr<<<CDIV(E2v,256),256,0,stream>>>(d2, cursor, eid, E2v);

  // ---- stage 2 GAT stack (relu on all three layers; layer 0 has K=384) ----
  gat_layer(stream, xh_a, H3, N1v, E2v, s2, d2, m2,
            gc_W[1][0], gc_as[1][0], gc_ad[1][0], gc_b[1][0],
            1, gc_bn_w[1][0], gc_bn_b[1][0],
            wh, xp, asrc, adst, menc, ssum, alpha, rowstart, eid, bnsum,
            xall + 0*HH, xh_b, o_fea2, K1, fsc);
  for (int l=1;l<3;l++)
    gat_layer(stream, xh_b, HH, N1v, E2v, s2, d2, m2,
              gc_W[1][l], gc_as[1][l], gc_ad[1][l], gc_b[1][l],
              1, gc_bn_w[1][l], gc_bn_b[1][l],
              wh, xp, asrc, adst, menc, ssum, alpha, rowstart, eid, bnsum,
              xall + l*HH, xh_b, o_fea2, K1, fsc);

  // ---- TopK pool 2 + graph norm + global mean pool ----
  k_vnorm<<<1,512,0,stream>>>(pw2, H3, wnorm);
  k_score<<<CDIV(N1v,8),256,0,stream>>>(xall, pw2, wnorm, score_all, N1v);
  k_topk<K1,K2><<<NB,K1,0,stream>>>(score_all, perm2_i, o_perm2, o_score2);
  k_pool_gather<<<N2v,H3,0,stream>>>(xall, perm2_i, o_score2, xp1);     // xp1 reused as xp2
  k_gnorm<<<NB,H3,0,stream>>>(xp1, gn_w[1], gn_b[1], gn_ms[1], K2, xq, nullptr);
  k_gmean<<<NB,H3,0,stream>>>(xq, xg);

  // ---- MLP head (tiny; plain f32 for exactness) ----
  k_linrelu<<<CDIV(64*512,256),256,0,stream>>>(xg, W1, b1, h1, 64, 512, H3);
  k_bnsmall<<<CDIV(512,128),128,0,stream>>>(h1, bn1w, bn1b, h1, 64, 512);
  k_linrelu<<<CDIV(64*256,256),256,0,stream>>>(h1, W2, b2, h2, 64, 256, 512);
  k_bnsmall<<<CDIV(256,128),128,0,stream>>>(h2, bn2w, bn2b, o_h, 64, 256);
}
